// S4CondBlock_90735479096059
// MI455X (gfx1250) — matrix operand order repaired and checked by reading, not compiled, not run on hardware
//
#include <hip/hip_runtime.h>

typedef __attribute__((ext_vector_type(16))) _Float16 v16h;
typedef __attribute__((ext_vector_type(8)))  float    v8f;
typedef __attribute__((ext_vector_type(4)))  unsigned su4;
typedef __attribute__((ext_vector_type(8)))  unsigned su8;

static constexpr int HH  = 64;     // heads
static constexpr int BBATCH = 16;  // batch
static constexpr int LL  = 16384;  // sequence
static constexpr int NN  = 64;     // states
static constexpr int MCH = 64;     // chunk length
static constexpr int NCHUNK = LL / MCH;   // 256
static constexpr int NMATS = 5;    // T, Qr, -Qi, Pr, Pi

// workspace layout (bytes)
static constexpr size_t WS_PACK   = 0;
static constexpr size_t PACK_BYTES = (size_t)HH * NMATS * 4 * 2 * 32 * 16 * 2; // 2.62 MB
static constexpr size_t WS_LAMM   = WS_PACK + PACK_BYTES;
static constexpr size_t LAMM_BYTES = (size_t)HH * NN * 2 * 4;                  // 32 KB
static constexpr size_t WS_GB     = WS_LAMM + LAMM_BYTES;
static constexpr size_t GB_BYTES  = (size_t)BBATCH * 2 * HH * 4;               // 8 KB
static constexpr size_t WS_Y      = WS_GB + GB_BYTES;                          // + 64 MB y

// index into the A-operand pack: 16-bit A-matrix (16x32), contiguous k order (k = 16*half + elem)
__device__ __forceinline__ size_t pack_idx(int h, int mat, int m, int kcol) {
    int mt = m >> 4, ml = m & 15;
    int ks = kcol >> 5, kk = kcol & 31;
    int lane = ml + ((kk >> 4) << 4);
    int elem = kk & 15;
    return ((size_t)((((h * NMATS + mat) * 4 + mt) * 2 + ks) * 32 + lane)) * 16 + elem;
}

// ---------------------------------------------------------------------------
// TDM: DMA a [BBATCH rows x MCH cols] f32 tile (row stride HH*LL elements)
// from global memory into LDS at lds_off. ISA 08_async_tensor D# layout:
//   group0: count=1 | lds_addr | global_addr[56:0] | type=2
//   group1: data_size=4B | tensor_dim0=LL | tensor_dim1=BBATCH |
//           tile_dim0=MCH | tile_dim1=BBATCH | tensor_dim0_stride=HH*LL
// ---------------------------------------------------------------------------
__device__ __forceinline__ void tdm_load_tile(const float* gptr, unsigned lds_off) {
    unsigned long long ga = (unsigned long long)(size_t)gptr;
    su4 g0;
    g0[0] = 1u;                                               // count=1, user desc
    g0[1] = lds_off;                                          // LDS byte address
    g0[2] = (unsigned)(ga & 0xFFFFFFFFu);                     // global_addr lo
    g0[3] = (unsigned)((ga >> 32) & 0x01FFFFFFu) | (2u << 30);// addr[56:32] | type=2
    su8 g1;
    g1[0] = (2u << 16);                                       // data_size=2 -> 4 bytes
    g1[1] = ((unsigned)LL & 0xFFFFu) << 16;                   // tensor_dim0[15:0] @bits63:48
    g1[2] = (((unsigned)LL >> 16) & 0xFFFFu)                  // tensor_dim0[31:16]
          | (((unsigned)BBATCH & 0xFFFFu) << 16);             // tensor_dim1[15:0]
    g1[3] = (((unsigned)BBATCH >> 16) & 0xFFFFu)              // tensor_dim1[31:16] (=0)
          | (((unsigned)MCH & 0xFFFFu) << 16);                // tile_dim0 @bits127:112
    g1[4] = (unsigned)BBATCH;                                 // tile_dim1 (tile_dim2=0)
    g1[5] = (unsigned)(HH * LL);                              // tensor_dim0_stride lo32
    g1[6] = 0u;                                               // stride hi16 | dim1_stride lo
    g1[7] = 0u;
    asm volatile("tensor_load_to_lds %0, %1" : : "s"(g0), "s"(g1) : "memory");
}

// ---------------------------------------------------------------------------
// Kernel 1: per-head tables: Toeplitz T of K, cross matrices Q, state matrices
// P, and lambda^M — packed into WMMA A-operand layout.
// ---------------------------------------------------------------------------
__global__ void s4_tables(const float* __restrict__ log_dt,
                          const float* __restrict__ C_ri,
                          const float* __restrict__ log_A_real,
                          const float* __restrict__ A_imag,
                          _Float16* __restrict__ pack,
                          float* __restrict__ lamM) {
    __shared__ float pr[NN][MCH + 1];
    __shared__ float qi_[NN][MCH + 1];
    __shared__ float ckr[NN], cki[NN], kker[MCH];
    const int h = blockIdx.x;
    const int t = threadIdx.x;            // 0..63
    const float dt = expf(log_dt[h]);
    {
        int n = t;
        float Ar = -expf(log_A_real[h * NN + n]);
        float Ai = A_imag[h * NN + n];
        float dr = dt * Ar, di = dt * Ai;
        for (int j = 0; j <= MCH; ++j) {
            float e = expf(dr * (float)j);
            pr[n][j]  = e * cosf(di * (float)j);
            qi_[n][j] = e * sinf(di * (float)j);
        }
        float l1r = pr[n][1] - 1.f, l1i = qi_[n][1];
        float den = Ar * Ar + Ai * Ai;
        float tr = (l1r * Ar + l1i * Ai) / den;
        float ti = (l1i * Ar - l1r * Ai) / den;
        float Cr = C_ri[(h * NN + n) * 2 + 0];
        float Ci = C_ri[(h * NN + n) * 2 + 1];
        ckr[n] = Cr * tr - Ci * ti;
        cki[n] = Cr * ti + Ci * tr;
        lamM[(h * NN + n) * 2 + 0] = pr[n][MCH];
        lamM[(h * NN + n) * 2 + 1] = qi_[n][MCH];
    }
    __syncthreads();
    {
        int j = t;
        float s = 0.f;
        for (int n = 0; n < NN; ++n) s += ckr[n] * pr[n][j] - cki[n] * qi_[n][j];
        kker[j] = 2.f * s;
    }
    __syncthreads();
    {
        int j = t;
        for (int c = 0; c < MCH; ++c)
            pack[pack_idx(h, 0, j, c)] = (_Float16)((c <= j) ? kker[j - c] : 0.f);
        for (int n = 0; n < NN; ++n) {
            float qr = 2.f * (ckr[n] * pr[n][j + 1] - cki[n] * qi_[n][j + 1]);
            float qi = 2.f * (ckr[n] * qi_[n][j + 1] + cki[n] * pr[n][j + 1]);
            pack[pack_idx(h, 1, j, n)] = (_Float16)qr;
            pack[pack_idx(h, 2, j, n)] = (_Float16)(-qi);
        }
        int n = t;
        for (int c = 0; c < MCH; ++c) {
            pack[pack_idx(h, 3, n, c)] = (_Float16)pr[n][MCH - 1 - c];
            pack[pack_idx(h, 4, n, c)] = (_Float16)qi_[n][MCH - 1 - c];
        }
    }
}

// ---------------------------------------------------------------------------
// Kernel 2: FiLM gains/biases: gb[b][2H] = cond[b] @ film_w.T + film_b
// ---------------------------------------------------------------------------
__global__ void s4_film(const float* __restrict__ cond,
                        const float* __restrict__ film_w,
                        const float* __restrict__ film_b,
                        float* __restrict__ gbv) {
    int i = blockIdx.x * blockDim.x + threadIdx.x;
    if (i < BBATCH * 2 * HH) {
        int b = i >> 7, j = i & 127;
        float s = film_b[j];
        for (int c = 0; c < 3; ++c) s += cond[b * 3 + c] * film_w[j * 3 + c];
        gbv[b * 128 + j] = s;
    }
}

#define WMMA_F16(A, Bm, C) \
    __builtin_amdgcn_wmma_f32_16x16x32_f16(false, (A), false, (Bm), (short)0, (C), false, false)

// ---------------------------------------------------------------------------
// Kernel 3: chunked WMMA scan. TDM double-buffers the u tile into LDS one
// chunk ahead; waves 0..3 produce y tiles, waves 4..7 carry the SSM state.
// ---------------------------------------------------------------------------
__global__ void __launch_bounds__(256)
s4_scan(const float* __restrict__ u,
        const int* __restrict__ length,
        const float* __restrict__ Dvec,
        const _Float16* __restrict__ pack,
        const float* __restrict__ lamM,
        const float* __restrict__ gbv,
        float* __restrict__ y_ws,
        float* __restrict__ out) {
    __shared__ __align__(16) float sRaw[2][BBATCH][MCH];     // TDM landing pad
    __shared__ __align__(16) _Float16 sU[2][32][16];         // B-operand swizzled U
    __shared__ float sUm[MCH][BBATCH];                       // masked u, f32
    __shared__ __align__(16) _Float16 sXh[2][2][2][32][16];  // state f16 B-operand
    __shared__ float sXf[2][2][NN][BBATCH];                  // state f32
    __shared__ float sLam[NN][2];
    __shared__ int   sLen[BBATCH];
    __shared__ float sGB[BBATCH][2];
    __shared__ float sRed[2];
    __shared__ float sMeanInv[2];

    const int h    = blockIdx.x;
    const int tid  = threadIdx.x;
    const int wave = tid >> 5;
    const int lane = tid & 31;
    const int hi   = lane >> 4;
    const int bcol = lane & 15;
    const float Dh = Dvec[h];

    if (tid < BBATCH) {
        sLen[tid]    = length[tid];
        sGB[tid][0]  = gbv[tid * 128 + h];       // gamma
        sGB[tid][1]  = gbv[tid * 128 + 64 + h];  // beta
    }
    if (tid < 128) sLam[tid >> 1][tid & 1] = lamM[h * 128 + tid];
    if (tid < 2) sRed[tid] = 0.f;
    for (int i = tid; i < 2 * NN * BBATCH; i += 256) {
        ((float*)sXf[0])[i]    = 0.f;
        ((_Float16*)sXh[0])[i] = (_Float16)0.f;
    }

    // A-operand fragments held in registers for the whole scan
    v16h aT[2], aQr[2], aQn[2];   // y waves
    v16h aPr[2], aPi[2];          // state waves
    if (wave < 4) {
        const int mt = wave;
        for (int ks = 0; ks < 2; ++ks) {
            aT[ks]  = *(const v16h*)(pack + ((size_t)((((h * NMATS + 0) * 4 + mt) * 2 + ks) * 32 + lane)) * 16);
            aQr[ks] = *(const v16h*)(pack + ((size_t)((((h * NMATS + 1) * 4 + mt) * 2 + ks) * 32 + lane)) * 16);
            aQn[ks] = *(const v16h*)(pack + ((size_t)((((h * NMATS + 2) * 4 + mt) * 2 + ks) * 32 + lane)) * 16);
        }
    } else {
        const int mt = wave - 4;
        for (int ks = 0; ks < 2; ++ks) {
            aPr[ks] = *(const v16h*)(pack + ((size_t)((((h * NMATS + 3) * 4 + mt) * 2 + ks) * 32 + lane)) * 16);
            aPi[ks] = *(const v16h*)(pack + ((size_t)((((h * NMATS + 4) * 4 + mt) * 2 + ks) * 32 + lane)) * 16);
        }
    }

    // loader mapping: thread -> (batch, time-quad) within the chunk
    const int bld = tid >> 4;
    const int tq  = (tid & 15) << 2;
    const float* ubase = u + ((size_t)h << 14);          // row b=0 of this head
    const unsigned ldsRaw0 = (unsigned)(size_t)&sRaw[0][0][0];
    const unsigned ldsRaw1 = (unsigned)(size_t)&sRaw[1][0][0];

    // prime the TDM pipeline: chunk 0 -> buffer 0
    if (wave == 0) tdm_load_tile(ubase, ldsRaw0);
    __syncthreads();

    float s1 = 0.f, s2 = 0.f;

    for (int c = 0; c < NCHUNK; ++c) {
        const int cur = c & 1, nxt = cur ^ 1;
        // issue next chunk's DMA, wait for current chunk's DMA (in-order)
        if (wave == 0) {
            if (c + 1 < NCHUNK) {
                tdm_load_tile(ubase + (size_t)(c + 1) * MCH, nxt ? ldsRaw1 : ldsRaw0);
                __builtin_amdgcn_s_wait_tensorcnt(1);
            } else {
                __builtin_amdgcn_s_wait_tensorcnt(0);
            }
        }
        __syncthreads();   // raw tile `cur` visible; previous consumers done

        // ---- mask + cvt + swizzle into WMMA B-operand layout ----
        {
            const float4 rv = *(const float4*)&sRaw[cur][bld][tq];
            const int t0 = c * MCH + tq;
            const int len = sLen[bld];
            float fv[4] = { (t0 + 0 < len) ? rv.x : 0.f,
                            (t0 + 1 < len) ? rv.y : 0.f,
                            (t0 + 2 < len) ? rv.z : 0.f,
                            (t0 + 3 < len) ? rv.w : 0.f };
            for (int e = 0; e < 4; ++e) {
                int tt = tq + e;
                sUm[tt][bld] = fv[e];
                int ks = tt >> 5, r = tt & 31;
                sU[ks][bld + ((r >> 4) << 4)][r & 15] = (_Float16)fv[e];
            }
        }
        __syncthreads();   // U chunk ready

        const v16h bU0 = *(const v16h*)&sU[0][lane][0];
        const v16h bU1 = *(const v16h*)&sU[1][lane][0];

        if (wave < 4) {
            const v16h bXr0 = *(const v16h*)&sXh[cur][0][0][lane][0];
            const v16h bXr1 = *(const v16h*)&sXh[cur][0][1][lane][0];
            const v16h bXi0 = *(const v16h*)&sXh[cur][1][0][lane][0];
            const v16h bXi1 = *(const v16h*)&sXh[cur][1][1][lane][0];
            v8f acc = {};
            acc = WMMA_F16(aT[0],  bU0,  acc);
            acc = WMMA_F16(aT[1],  bU1,  acc);
            acc = WMMA_F16(aQr[0], bXr0, acc);
            acc = WMMA_F16(aQr[1], bXr1, acc);
            acc = WMMA_F16(aQn[0], bXi0, acc);
            acc = WMMA_F16(aQn[1], bXi1, acc);
            const int mylen = sLen[bcol];
            const int jbase = (wave << 4) + (hi << 3);
            float vals[8];
            for (int k = 0; k < 8; ++k) {
                int j = jbase + k;
                float yv = acc[k] + sUm[j][bcol] * Dh;
                if (c * MCH + j >= mylen) yv = 0.f;
                vals[k] = yv;
                s1 += yv;
                s2 += yv * yv;
            }
            float* yp = y_ws + (((size_t)(h * BBATCH + bcol)) << 14) + c * MCH + jbase;
            float4 lo; lo.x = vals[0]; lo.y = vals[1]; lo.z = vals[2]; lo.w = vals[3];
            float4 hi4; hi4.x = vals[4]; hi4.y = vals[5]; hi4.z = vals[6]; hi4.w = vals[7];
            *(float4*)(yp)     = lo;
            *(float4*)(yp + 4) = hi4;
        } else {
            v8f gr = {}, gi = {};
            gr = WMMA_F16(aPr[0], bU0, gr);
            gr = WMMA_F16(aPr[1], bU1, gr);
            gi = WMMA_F16(aPi[0], bU0, gi);
            gi = WMMA_F16(aPi[1], bU1, gi);
            const int nbase = ((wave - 4) << 4) + (hi << 3);
            for (int k = 0; k < 8; ++k) {
                int n = nbase + k;
                float xr = sXf[cur][0][n][bcol];
                float xi = sXf[cur][1][n][bcol];
                float lr = sLam[n][0], li = sLam[n][1];
                float nr = lr * xr - li * xi + gr[k];
                float ni = li * xr + lr * xi + gi[k];
                sXf[nxt][0][n][bcol] = nr;
                sXf[nxt][1][n][bcol] = ni;
                int ks = n >> 5, r = n & 31;
                int l2 = bcol + ((r >> 4) << 4), el = r & 15;
                sXh[nxt][0][ks][l2][el] = (_Float16)nr;
                sXh[nxt][1][ks][l2][el] = (_Float16)ni;
            }
        }
    }

    // ---- per-head mean / var over (B, L) ----
    __syncthreads();
    if (wave < 4) {
        atomicAdd(&sRed[0], s1);
        atomicAdd(&sRed[1], s2);
    }
    __syncthreads();
    if (tid == 0) {
        const float cnt = (float)(BBATCH * LL);
        float mean = sRed[0] / cnt;
        float var  = sRed[1] / cnt - mean * mean;
        sMeanInv[0] = mean;
        sMeanInv[1] = rsqrtf(var + 1e-5f);
    }
    __syncthreads();
    const float mean = sMeanInv[0];
    const float inv  = sMeanInv[1];

    // ---- pass 2: FiLM + tanh + residual ----
    for (int i4 = tid; i4 < (BBATCH * LL) / 4; i4 += 256) {
        int f = i4 << 2;
        int b = f >> 14, l = f & (LL - 1);
        const float4 yv = *(const float4*)(y_ws + (((size_t)(h * BBATCH + b)) << 14) + l);
        const float4 u4 = *(const float4*)(u + (((size_t)(b * HH + h)) << 14) + l);
        const float g = sGB[b][0], be = sGB[b][1];
        float4 o;
        o.x = u4.x + tanhf((yv.x - mean) * inv * g + be);
        o.y = u4.y + tanhf((yv.y - mean) * inv * g + be);
        o.z = u4.z + tanhf((yv.z - mean) * inv * g + be);
        o.w = u4.w + tanhf((yv.w - mean) * inv * g + be);
        *(float4*)(out + (((size_t)(b * HH + h)) << 14) + l) = o;
    }
}

extern "C" void kernel_launch(void* const* d_in, const int* in_sizes, int n_in,
                              void* d_out, int out_size, void* d_ws, size_t ws_size,
                              hipStream_t stream) {
    (void)in_sizes; (void)n_in; (void)out_size; (void)ws_size;
    const float* u       = (const float*)d_in[0];
    const float* cond    = (const float*)d_in[1];
    const int*   length  = (const int*)d_in[2];
    const float* log_dt  = (const float*)d_in[3];
    const float* C_ri    = (const float*)d_in[4];
    const float* log_A   = (const float*)d_in[5];
    const float* A_imag  = (const float*)d_in[6];
    const float* Dv      = (const float*)d_in[7];
    const float* film_w  = (const float*)d_in[8];
    const float* film_b  = (const float*)d_in[9];

    char* ws = (char*)d_ws;
    _Float16* pack = (_Float16*)(ws + WS_PACK);
    float*    lamM = (float*)(ws + WS_LAMM);
    float*    gbv  = (float*)(ws + WS_GB);
    float*    y_ws = (float*)(ws + WS_Y);
    float*    out  = (float*)d_out;

    s4_tables<<<HH, 64, 0, stream>>>(log_dt, C_ri, log_A, A_imag, pack, lamM);
    s4_film<<<(BBATCH * 2 * HH + 255) / 256, 256, 0, stream>>>(cond, film_w, film_b, gbv);
    s4_scan<<<HH, 256, 0, stream>>>(u, length, Dv, pack, lamM, gbv, y_ws, out);
}
